// CustomCrossAttentionMinimal_38646115729799
// MI455X (gfx1250) — matrix operand order repaired and checked by reading, not compiled, not run on hardware
//
#include <hip/hip_runtime.h>
#include <hip/hip_bf16.h>
#include <math.h>

// ---------------------------------------------------------------------------
// Problem constants (from reference)
// ---------------------------------------------------------------------------
#define Bx     16
#define SEQ    4096
#define NCTX   77
#define NH     8
#define DHD    40
#define QD     320
#define CD     768
#define INNER  320
#define ATT_SCALE 0.15811388300841897f   // 40^-0.5
#define NEGBIG   -3.4028235e38f

typedef __attribute__((ext_vector_type(16))) _Float16 v16h;
typedef __attribute__((ext_vector_type(8)))  _Float16 v8h;
typedef __attribute__((ext_vector_type(4)))  _Float16 v4h;
typedef __attribute__((ext_vector_type(8)))  float    v8f;

// ---------------------------------------------------------------------------
// Generic fp32-in / fp32-out GEMM, f16 WMMA compute, fp32 accumulate.
// C[M,N] = A[M,K] (row) @ B[K,N] (row) (+ bias[n])
// Block tile 128x64, K-step 32, 256 threads = 8 waves, wave tile 32x32.
// A staged [M][K] halves; B staged TRANSPOSED [N][K] halves so both
// fragment gathers are contiguous 8-half runs -> ds_load_b128 pairs.
// ---------------------------------------------------------------------------
#define BM 128
#define BN 64
#define BK 32
#define LDA  40   // halves stride for A tile rows (pad 32 -> 40)
#define LDBT 40   // halves stride for B^T tile rows (pad 32 -> 40)

__global__ __launch_bounds__(256)
void gemm_wmma_f16(const float* __restrict__ A, const float* __restrict__ B,
                   const float* __restrict__ bias, float* __restrict__ C,
                   int M, int N, int K)
{
    __shared__ __align__(16) _Float16 lA [BM * LDA];
    __shared__ __align__(16) _Float16 lBt[BN * LDBT];

    const int tid  = threadIdx.x;
    const int lane = tid & 31;
    const int wave = tid >> 5;
    const int wm   = (wave & 3) * 32;   // wave M offset in tile
    const int wn   = (wave >> 2) * 32;  // wave N offset in tile
    const int blockM = blockIdx.x * BM;
    const int blockN = blockIdx.y * BN;

    v8f acc[2][2];
#pragma unroll
    for (int i = 0; i < 2; ++i)
#pragma unroll
        for (int j = 0; j < 2; ++j) {
            v8f z = {0.f, 0.f, 0.f, 0.f, 0.f, 0.f, 0.f, 0.f};
            acc[i][j] = z;
        }

    // Per-thread cooperative tile-load coordinates
    const int arow  = (tid * 16) >> 5;   // A: 16 halves/thread, rows 0..127
    const int acol  = (tid * 16) & 31;   // 0 or 16
    const int bcolT = tid & 63;          // B^T: column within tile
    const int bk0   = (tid >> 6) * 8;    // B^T: K start (8 K-values/thread)

    for (int kt = 0; kt < K; kt += BK) {
        // ---- stage A tile [M][K] halves (guard M edge) ----
        {
            const int gr = blockM + arow;
            if (gr < M) {
                const float4* src = (const float4*)(A + (size_t)gr * K + kt + acol);
#pragma unroll
                for (int u = 0; u < 4; ++u) {
                    float4 f = src[u];
                    v4h hv;
                    hv[0] = (_Float16)f.x; hv[1] = (_Float16)f.y;
                    hv[2] = (_Float16)f.z; hv[3] = (_Float16)f.w;
                    *(v4h*)&lA[arow * LDA + acol + u * 4] = hv;
                }
                // prefetch next K-step of this A row (global_prefetch_b8)
                if (kt + BK < K)
                    __builtin_prefetch(A + (size_t)gr * K + kt + BK + acol, 0, 3);
            } else {
                v4h zz; zz[0] = zz[1] = zz[2] = zz[3] = (_Float16)0.0f;
#pragma unroll
                for (int u = 0; u < 4; ++u)
                    *(v4h*)&lA[arow * LDA + acol + u * 4] = zz;
            }
        }
        // ---- stage B tile TRANSPOSED [N][K] halves ----
        // lanes cover consecutive columns -> each strided b32 load coalesces.
        {
            const float* src = B + (size_t)(kt + bk0) * N + blockN + bcolT;
            float f[8];
#pragma unroll
            for (int u = 0; u < 8; ++u) f[u] = src[(size_t)u * N];
#pragma unroll
            for (int u = 0; u < 2; ++u) {
                v4h hv;
                hv[0] = (_Float16)f[u * 4 + 0];
                hv[1] = (_Float16)f[u * 4 + 1];
                hv[2] = (_Float16)f[u * 4 + 2];
                hv[3] = (_Float16)f[u * 4 + 3];
                *(v4h*)&lBt[bcolT * LDBT + bk0 + u * 4] = hv;
            }
        }
        __syncthreads();

        // ---- gather WMMA fragments (ISA 7.12.2 16-bit layouts) ----
        // A 16x32: lanes0-15 M=lane (K 0..7 | 16..23); lanes16-31 (K 8..15 | 24..31)
        // B 32x16: lanes0-15 N=lane (K 0..7 | 16..23); lanes16-31 (K 8..15 | 24..31)
        v16h af[2], bf[2];
#pragma unroll
        for (int t = 0; t < 2; ++t) {
            const int row = wm + t * 16 + (lane & 15);
            const int kb  = (lane >> 4) * 8;
            v8h alo = *(const v8h*)&lA[row * LDA + kb];
            v8h ahi = *(const v8h*)&lA[row * LDA + kb + 16];
            af[t] = __builtin_shufflevector(alo, ahi,
                    0, 1, 2, 3, 4, 5, 6, 7, 8, 9, 10, 11, 12, 13, 14, 15);

            const int col = wn + t * 16 + (lane & 15);
            const int kb2 = (lane >> 4) * 8;
            v8h blo = *(const v8h*)&lBt[col * LDBT + kb2];
            v8h bhi = *(const v8h*)&lBt[col * LDBT + kb2 + 16];
            bf[t] = __builtin_shufflevector(blo, bhi,
                    0, 1, 2, 3, 4, 5, 6, 7, 8, 9, 10, 11, 12, 13, 14, 15);
        }

#pragma unroll
        for (int tm = 0; tm < 2; ++tm)
#pragma unroll
            for (int tn = 0; tn < 2; ++tn)
                acc[tm][tn] = __builtin_amdgcn_wmma_f32_16x16x32_f16(
                    false, af[tm], false, bf[tn], (short)0, acc[tm][tn],
                    false, false);
        __syncthreads();
    }

    // ---- store C (lanes0-15: N=lane, M=r; lanes16-31: M=8+r) ----
#pragma unroll
    for (int tm = 0; tm < 2; ++tm) {
        const int row0 = blockM + wm + tm * 16 + ((lane >> 4) << 3);
#pragma unroll
        for (int tn = 0; tn < 2; ++tn) {
            const int col = blockN + wn + tn * 16 + (lane & 15);
            const float bv = bias ? bias[col] : 0.0f;
#pragma unroll
            for (int r = 0; r < 8; ++r) {
                const int row = row0 + r;
                if (row < M)
                    C[(size_t)row * N + col] = acc[tm][tn][r] + bv;
            }
        }
    }
}

// ---------------------------------------------------------------------------
// Progress classifier: e = emb1[xs/20] + emb2[xs%20];
// h = gelu(gelu(e) @ Wc1 + bc1); logits = h @ Wc2 + bc2; pairwise softmax.
// ---------------------------------------------------------------------------
__device__ inline float gelu_erf(float x) {
    return 0.5f * x * (1.0f + erff(x * 0.7071067811865476f));
}

__global__ void pc_kernel(const float* __restrict__ progress,
                          const float* __restrict__ emb1, const float* __restrict__ emb2,
                          const float* __restrict__ Wc1, const float* __restrict__ bc1,
                          const float* __restrict__ Wc2, const float* __restrict__ bc2,
                          float* __restrict__ pcw)
{
    __shared__ float g[512];
    __shared__ float h1[512];
    __shared__ float lg[16];
    const int tid = threadIdx.x;
    for (int b = 0; b < Bx; ++b) {
        int xs = (int)roundf(progress[b] * 1000.0f);
        if (xs > 999) xs = 999;
        const float* e1 = emb1 + (size_t)(xs / 20) * 512;
        const float* e2 = emb2 + (size_t)(xs % 20) * 512;
        for (int c = tid; c < 512; c += blockDim.x)
            g[c] = gelu_erf(e1[c] + e2[c]);
        __syncthreads();
        for (int c = tid; c < 512; c += blockDim.x) {
            float s = bc1[c];
            for (int k = 0; k < 512; ++k) s += g[k] * Wc1[k * 512 + c];
            h1[c] = gelu_erf(s);
        }
        __syncthreads();
        if (tid < 16) {
            float s = bc2[tid];
            for (int k = 0; k < 512; ++k) s += h1[k] * Wc2[k * 16 + tid];
            lg[tid] = s;
        }
        __syncthreads();
        if (tid < NH) {
            const float l0 = lg[2 * tid], l1 = lg[2 * tid + 1];
            const float m  = fmaxf(l0, l1);
            const float e0 = expf(l0 - m), ee1 = expf(l1 - m);
            const float den = e0 + ee1;
            pcw[b * 16 + 2 * tid + 0] = e0 / den;
            pcw[b * 16 + 2 * tid + 1] = ee1 / den;
        }
        __syncthreads();
    }
}

// ---------------------------------------------------------------------------
// stats[0..3] = 0
// ---------------------------------------------------------------------------
__global__ void init_stats(float* __restrict__ stats) {
    if (threadIdx.x < 4) stats[threadIdx.x] = 0.0f;
}

// ---------------------------------------------------------------------------
// Pass 1: sum & sumsq of sim over captiontype>=2 columns (all b,h,i rows).
// ---------------------------------------------------------------------------
__global__ __launch_bounds__(256)
void sim_stats_kernel(const float* __restrict__ qg, const float* __restrict__ kg,
                      const int* __restrict__ ctypes, float* __restrict__ stats)
{
    __shared__ float sk[NCTX * DHD];
    __shared__ int   sct[NCTX];
    __shared__ float r1[256], r2[256];
    const int tid = threadIdx.x;
    const int h = blockIdx.y;
    const int b = blockIdx.z;

    for (int idx = tid; idx < NCTX * DHD; idx += 256) {
        const int j = idx / DHD, d = idx % DHD;
        sk[idx] = kg[((size_t)b * NCTX + j) * INNER + h * DHD + d];
    }
    for (int j = tid; j < NCTX; j += 256) sct[j] = ctypes[b * NCTX + j];
    __syncthreads();

    const int row = blockIdx.x * 256 + tid;
    const float* qp = qg + ((size_t)b * SEQ + row) * INNER + h * DHD;
    float qreg[DHD];
#pragma unroll
    for (int d = 0; d < DHD; ++d) qreg[d] = qp[d];

    float s1 = 0.f, s2 = 0.f;
    for (int j = 0; j < NCTX; ++j) {
        if (sct[j] >= 2) {
            float acc = 0.f;
#pragma unroll
            for (int d = 0; d < DHD; ++d) acc += qreg[d] * sk[j * DHD + d];
            s1 += acc;
            s2 += acc * acc;
        }
    }
    r1[tid] = s1; r2[tid] = s2;
    __syncthreads();
    for (int s = 128; s > 0; s >>= 1) {
        if (tid < s) { r1[tid] += r1[tid + s]; r2[tid] += r2[tid + s]; }
        __syncthreads();
    }
    if (tid == 0) {
        atomicAdd(&stats[0], r1[0]);
        atomicAdd(&stats[1], r2[0]);
    }
}

// ---------------------------------------------------------------------------
// Finalize: unbiased std over masked entries -> stats[2]
// ---------------------------------------------------------------------------
__global__ void finalize_stats(const int* __restrict__ ctypes, float* __restrict__ stats) {
    if (threadIdx.x == 0) {
        int cnt = 0;
        for (int t = 0; t < Bx * NCTX; ++t) cnt += (ctypes[t] >= 2) ? 1 : 0;
        const float n = (float)cnt * (float)NH * (float)SEQ;
        const float mean = stats[0] / n;
        float var = (stats[1] - n * mean * mean) / (n - 1.0f);
        stats[2] = sqrtf(fmaxf(var, 0.0f));
    }
}

// ---------------------------------------------------------------------------
// Pass 2: fused attention core. One thread = one query row.
// Recomputes sim (fp32), builds masks/wf, dual softmax, mixes with pc,
// multiplies by V, writes ctx[b,i,(h d)].
// ---------------------------------------------------------------------------
__global__ __launch_bounds__(64)
void attn_kernel(const float* __restrict__ qg, const float* __restrict__ kg,
                 const float* __restrict__ vg, const float* __restrict__ cam,
                 const int* __restrict__ ctypes, const float* __restrict__ pcw,
                 const float* __restrict__ stats, const float* __restrict__ strength,
                 float* __restrict__ ctx)
{
    __shared__ float sk[NCTX * DHD];
    __shared__ float sv[NCTX * DHD];
    __shared__ float ssim[64 * NCTX];  // stride 77 (coprime w/ 64 banks) -> conflict-free
    __shared__ int   sct[NCTX];
    const int tid = threadIdx.x;
    const int h = blockIdx.y;
    const int b = blockIdx.z;

    for (int idx = tid; idx < NCTX * DHD; idx += 64) {
        const int j = idx / DHD, d = idx % DHD;
        const size_t base = ((size_t)b * NCTX + j) * INNER + h * DHD + d;
        sk[idx] = kg[base];
        sv[idx] = vg[base];
    }
    for (int j = tid; j < NCTX; j += 64) sct[j] = ctypes[b * NCTX + j];
    __syncthreads();

    const int row = blockIdx.x * 64 + tid;
    const float* qp = qg + ((size_t)b * SEQ + row) * INNER + h * DHD;
    float qreg[DHD];
#pragma unroll
    for (int d = 0; d < DHD; ++d) qreg[d] = qp[d];

    float* mysim = &ssim[tid * NCTX];
    for (int j = 0; j < NCTX; ++j) {
        float acc = 0.f;
#pragma unroll
        for (int d = 0; d < DHD; ++d) acc += qreg[d] * sk[j * DHD + d];
        mysim[j] = acc;
    }

    // weight function wf = where(ct>=2, cam * std * strength, 0)
    const float c = stats[2] * strength[0];
    const float* camp = cam + ((size_t)b * SEQ + row) * NCTX;
    float wfmax = -3.4e38f, wfsum = 0.f;
    for (int j = 0; j < NCTX; ++j) {
        const float wfj = (sct[j] >= 2) ? camp[j] * c : 0.0f;
        wfsum += wfj;
        wfmax = fmaxf(wfmax, wfj);
    }
    const bool  useg   = (wfsum == 0.0f);
    const float thresh = fmaxf(wfmax, 0.001f) - 0.0001f;

    // local-mask bits (cam read #2 hits L2)
    unsigned lmb[3] = {0u, 0u, 0u};
    for (int j = 0; j < NCTX; ++j) {
        const int ct = sct[j];
        const float wfj = (ct >= 2) ? camp[j] * c : 0.0f;
        const bool gm = (ct >= 0) && (ct < 2);
        if ((wfj >= thresh) || (useg && gm))
            lmb[j >> 5] |= (1u << (j & 31));
    }

    // dual softmax: max pass
    float gmax = -3.4e38f, lmax = -3.4e38f;
    for (int j = 0; j < NCTX; ++j) {
        const int ct = sct[j];
        const bool gm = (ct >= 0) && (ct < 2);
        const bool lm = (lmb[j >> 5] >> (j & 31)) & 1u;
        const float s = mysim[j];
        gmax = fmaxf(gmax, (gm ? s : NEGBIG) * ATT_SCALE);
        lmax = fmaxf(lmax, (lm ? s : NEGBIG) * ATT_SCALE);
    }
    // denominators
    float gden = 0.f, lden = 0.f;
    for (int j = 0; j < NCTX; ++j) {
        const int ct = sct[j];
        const bool gm = (ct >= 0) && (ct < 2);
        const bool lm = (lmb[j >> 5] >> (j & 31)) & 1u;
        const float s = mysim[j];
        gden += expf((gm ? s : NEGBIG) * ATT_SCALE - gmax);
        lden += expf((lm ? s : NEGBIG) * ATT_SCALE - lmax);
    }
    const float pc0 = pcw[(b * NH + h) * 2 + 0] / gden;
    const float pc1 = pcw[(b * NH + h) * 2 + 1] / lden;

    float outr[DHD];
#pragma unroll
    for (int d = 0; d < DHD; ++d) outr[d] = 0.f;
    for (int j = 0; j < NCTX; ++j) {
        const int ct = sct[j];
        const bool gm = (ct >= 0) && (ct < 2);
        const bool lm = (lmb[j >> 5] >> (j & 31)) & 1u;
        const float s = mysim[j];
        const float aj = pc0 * expf((gm ? s : NEGBIG) * ATT_SCALE - gmax)
                       + pc1 * expf((lm ? s : NEGBIG) * ATT_SCALE - lmax);
#pragma unroll
        for (int d = 0; d < DHD; ++d) outr[d] += aj * sv[j * DHD + d];
    }
    float* op = ctx + ((size_t)b * SEQ + row) * INNER + h * DHD;
#pragma unroll
    for (int d = 0; d < DHD; ++d) op[d] = outr[d];
}

// ---------------------------------------------------------------------------
// Host launcher
// ---------------------------------------------------------------------------
extern "C" void kernel_launch(void* const* d_in, const int* in_sizes, int n_in,
                              void* d_out, int out_size, void* d_ws, size_t ws_size,
                              hipStream_t stream) {
    (void)in_sizes; (void)n_in; (void)out_size; (void)ws_size;

    const float* x        = (const float*)d_in[0];
    const float* embs     = (const float*)d_in[1];
    const float* cam      = (const float*)d_in[2];
    const float* progress = (const float*)d_in[3];
    const float* strength = (const float*)d_in[4];
    const float* Wq       = (const float*)d_in[5];
    const float* Wk       = (const float*)d_in[6];
    const float* Wv       = (const float*)d_in[7];
    const float* Wo       = (const float*)d_in[8];
    const float* bo       = (const float*)d_in[9];
    const float* emb1     = (const float*)d_in[10];
    const float* emb2     = (const float*)d_in[11];
    const float* Wc1      = (const float*)d_in[12];
    const float* bc1      = (const float*)d_in[13];
    const float* Wc2      = (const float*)d_in[14];
    const float* bc2      = (const float*)d_in[15];
    const int*   ctypes   = (const int*)d_in[16];
    float* out = (float*)d_out;

    float* q_ws   = (float*)d_ws;                        // 65536*320
    float* ctx_ws = q_ws   + (size_t)Bx * SEQ  * INNER;  // 65536*320
    float* k_ws   = ctx_ws + (size_t)Bx * SEQ  * INNER;  // 1232*320
    float* v_ws   = k_ws   + (size_t)Bx * NCTX * INNER;  // 1232*320
    float* pc_ws  = v_ws   + (size_t)Bx * NCTX * INNER;  // 256
    float* stats  = pc_ws  + (size_t)Bx * NH * 2;        // 4

    const int Mq  = Bx * SEQ;    // 65536
    const int Mkv = Bx * NCTX;   // 1232

    init_stats<<<1, 32, 0, stream>>>(stats);

    gemm_wmma_f16<<<dim3((Mq + BM - 1) / BM, INNER / BN), 256, 0, stream>>>(
        x, Wq, (const float*)nullptr, q_ws, Mq, INNER, QD);
    gemm_wmma_f16<<<dim3((Mkv + BM - 1) / BM, INNER / BN), 256, 0, stream>>>(
        embs, Wk, (const float*)nullptr, k_ws, Mkv, INNER, CD);
    gemm_wmma_f16<<<dim3((Mkv + BM - 1) / BM, INNER / BN), 256, 0, stream>>>(
        embs, Wv, (const float*)nullptr, v_ws, Mkv, INNER, CD);

    pc_kernel<<<1, 256, 0, stream>>>(progress, emb1, emb2, Wc1, bc1, Wc2, bc2, pc_ws);

    sim_stats_kernel<<<dim3(SEQ / 256, NH, Bx), 256, 0, stream>>>(q_ws, k_ws, ctypes, stats);
    finalize_stats<<<1, 32, 0, stream>>>(ctypes, stats);

    attn_kernel<<<dim3(SEQ / 64, NH, Bx), 64, 0, stream>>>(
        q_ws, k_ws, v_ws, cam, ctypes, pc_ws, stats, strength, ctx_ws);

    gemm_wmma_f16<<<dim3(Mq / BM, QD / BN), 256, 0, stream>>>(
        ctx_ws, Wo, bo, out, Mq, QD, INNER);
}
